// VRNN_82781199663452
// MI455X (gfx1250) — compile-verified, matrix operand order checked by reading, not run on hardware
//
#include <hip/hip_runtime.h>
#include <math.h>

typedef __bf16 bf16_t;
typedef __attribute__((ext_vector_type(16))) __bf16 v16bf;
typedef __attribute__((ext_vector_type(8)))  float  v8f;

union FragAB { v16bf v; uint4 q[2]; };

#define ACT_NONE 0
#define ACT_RELU 1
#define ACT_SP   2

// ---------------------------------------------------------------------------
// bf16 WMMA GEMM, register double-buffered (all-constant indexing so fragments
// live in VGPRs), dual-problem batched on gridDim.z.
//   C[M,N] = act( concat(A0,A1)[M,K] @ Wt[N,K]^T + bias )
// Tile: block = 128(M) x 64(N), 8 waves as 4(M) x 2(N); each wave owns a
// 32x32 sub-tile (2 A-frags x 2 B-frags -> 4 v8f accumulators).
// Requires M%128==0, N%64==0, K%64==0 (true for every layer here).
// ---------------------------------------------------------------------------
__global__ __launch_bounds__(256)
void gemm_bf16_wmma(const bf16_t* __restrict__ A0a, const bf16_t* __restrict__ A0b,
                    long lda0,
                    const bf16_t* __restrict__ A1, long lda1, int K0,
                    const bf16_t* __restrict__ WtA, const bf16_t* __restrict__ WtB,
                    int K,
                    const float* __restrict__ biasA, const float* __restrict__ biasB,
                    float* __restrict__ CfA, float* __restrict__ CfB, long ldcf,
                    bf16_t* __restrict__ CbA, bf16_t* __restrict__ CbB, long ldcb,
                    int actA, int actB)
{
    const bool setB = (blockIdx.z != 0);
    const bf16_t* A0   = setB ? A0b   : A0a;
    const bf16_t* Wt   = setB ? WtB   : WtA;
    const float*  bias = setB ? biasB : biasA;
    float*        Cf   = setB ? CfB   : CfA;
    bf16_t*       Cb   = setB ? CbB   : CbA;
    const int     act  = setB ? actB  : actA;

    const int lane = threadIdx.x & 31;
    const int wave = threadIdx.x >> 5;
    const int wm   = wave & 3;       // 4 waves along M
    const int wn   = wave >> 2;      // 2 waves along N
    const int half = lane >> 4;      // K-half selector (ISA 16-bit A layout)
    const int lm   = lane & 15;

    const long mBase = (long)blockIdx.y * 128 + wm * 32;
    const int  nBase = blockIdx.x * 64 + wn * 32;

    long rowA[2];
    rowA[0] = mBase + lm;
    rowA[1] = mBase + 16 + lm;
    long colB[2];
    colB[0] = nBase + lm;
    colB[1] = nBase + 16 + lm;

    v8f acc[2][2] = {};
    FragAB a0[2], b0[2], a1[2], b1[2];   // two pipeline stages, named (no runtime idx)

    auto load_chunk = [&](int kb, FragAB (&a)[2], FragAB (&b)[2]) {
        const bf16_t* Ap; long lda; int koff;
        if (kb < K0) { Ap = A0; lda = lda0; koff = kb; }
        else         { Ap = A1; lda = lda1; koff = kb - K0; }
#pragma unroll
        for (int mi = 0; mi < 2; ++mi) {
            const bf16_t* ap = Ap + rowA[mi] * lda + koff + half * 8;
            a[mi].q[0] = *(const uint4*)(ap);
            a[mi].q[1] = *(const uint4*)(ap + 16);
        }
#pragma unroll
        for (int nj = 0; nj < 2; ++nj) {
            const bf16_t* bp = Wt + colB[nj] * (long)K + kb + half * 8;
            b[nj].q[0] = *(const uint4*)(bp);
            b[nj].q[1] = *(const uint4*)(bp + 16);
        }
    };
    auto do_wmma = [&](FragAB (&a)[2], FragAB (&b)[2]) {
#pragma unroll
        for (int mi = 0; mi < 2; ++mi)
#pragma unroll
            for (int nj = 0; nj < 2; ++nj)
                acc[mi][nj] = __builtin_amdgcn_wmma_f32_16x16x32_bf16(
                    false, a[mi].v, false, b[nj].v,
                    (short)0, acc[mi][nj], false, false);
    };

    load_chunk(0, a0, b0);
    load_chunk(32, a1, b1);
    for (int kb = 0; kb < K; kb += 64) {
        do_wmma(a0, b0);                               // chunk kb
        if (kb + 64 < K) load_chunk(kb + 64, a0, b0);  // refill, hidden by next group
        do_wmma(a1, b1);                               // chunk kb+32
        if (kb + 96 < K) load_chunk(kb + 96, a1, b1);
    }

    // C/D layout: lane -> N = lane%16 ; row-in-tile = vgpr r + (lane/16)*8
#pragma unroll
    for (int nj = 0; nj < 2; ++nj) {
        const int col = nBase + nj * 16 + lm;
        const float bb = bias ? bias[col] : 0.0f;
#pragma unroll
        for (int mi = 0; mi < 2; ++mi) {
            const long rowBase = mBase + mi * 16 + half * 8;
#pragma unroll
            for (int r = 0; r < 8; ++r) {
                float v = acc[mi][nj][r] + bb;
                if (act == ACT_RELU)    v = fmaxf(v, 0.0f);
                else if (act == ACT_SP) v = (v > 20.0f) ? v : log1pf(__expf(v));
                const long row = rowBase + r;
                if (Cf) Cf[row * ldcf + col] = v;
                if (Cb) Cb[row * ldcb + col] = (bf16_t)v;
            }
        }
    }
}

// Transpose + convert fp32 weight W[K,N] -> bf16 Wt[N,K]
__global__ void wcvt_kernel(const float* __restrict__ W, int K, int N,
                            bf16_t* __restrict__ Wt)
{
    long i = (long)blockIdx.x * blockDim.x + threadIdx.x;
    long total = (long)K * N;
    if (i >= total) return;
    int k = (int)(i / N);
    int n = (int)(i % N);
    Wt[(long)n * K + k] = (bf16_t)W[i];
}

__global__ void f2bf_kernel(const float* __restrict__ in, bf16_t* __restrict__ out, long n)
{
    long i = (long)blockIdx.x * blockDim.x + threadIdx.x;
    if (i < n) out[i] = (bf16_t)in[i];
}

__global__ void zerobf_kernel(bf16_t* __restrict__ p, long n)
{
    long i = (long)blockIdx.x * blockDim.x + threadIdx.x;
    if (i < n) p[i] = (bf16_t)0.0f;
}

// z_sample = z_mu + z_sigma * eps   (fp32 from d_out sections -> bf16)
__global__ void zsample_kernel(const float* __restrict__ zmu,
                               const float* __restrict__ zsig, long ld,
                               const float* __restrict__ eps, long lde,
                               bf16_t* __restrict__ zs, int n)
{
    int i = blockIdx.x * blockDim.x + threadIdx.x;
    if (i >= n) return;
    int b = i >> 8;        // Z = 256
    int c = i & 255;
    float v = zmu[(long)b * ld + c] + zsig[(long)b * ld + c] * eps[(long)b * lde + c];
    zs[i] = (bf16_t)v;
}

// GRUCell with hx==0: h = (1-z)*n ; gates from gi = inp@W_ih + b_ih and b_hh
__global__ void gru_gate_kernel(const float* __restrict__ gi,
                                const float* __restrict__ bhh,
                                bf16_t* __restrict__ h, int n)
{
    int i = blockIdx.x * blockDim.x + threadIdx.x;
    if (i >= n) return;
    int b = i >> 10;        // H = 1024
    int j = i & 1023;
    const float* g = gi + (long)b * 3072;
    float r  = 1.0f / (1.0f + __expf(-(g[j]        + bhh[j])));
    float z  = 1.0f / (1.0f + __expf(-(g[1024 + j] + bhh[1024 + j])));
    float nn = tanhf(g[2048 + j] + r * bhh[2048 + j]);
    h[i] = (bf16_t)((1.0f - z) * nn);
}

// ---------------------------------------------------------------------------

extern "C" void kernel_launch(void* const* d_in, const int* in_sizes, int n_in,
                              void* d_out, int out_size, void* d_ws, size_t ws_size,
                              hipStream_t stream)
{
    const int Bc = 128, Tt = 256, Xd = 512, Zd = 256, Hd = 1024;

    const float* x    = (const float*)d_in[0];     // [B,T,X]
    const float* eps  = (const float*)d_in[1];     // [B,T,Z]
    const float* b_hh = (const float*)d_in[57];

    struct WS { int idx, K, N; };
    enum { W_XF0, W_XF1, W_XF2, W_XF3, W_PF,
           W_PM0, W_PM1, W_PM2, W_PS0, W_PS1, W_PS2,
           W_EF,  W_EM0, W_EM1, W_EM2, W_ES0, W_ES1, W_ES2,
           W_ZF0, W_ZF1, W_ZF2, W_ZF3,
           W_D0,  W_D1,  W_D2,  W_D3,  W_IH, NW };
    const WS ws[NW] = {
        {2,512,1024},{4,1024,1024},{6,1024,1024},{8,1024,1024},
        {10,1024,1024},
        {12,1024,1024},{14,1024,1024},{16,1024,256},
        {18,1024,1024},{20,1024,1024},{22,1024,256},
        {24,2048,1024},
        {26,1024,1024},{28,1024,1024},{30,1024,256},
        {32,1024,1024},{34,1024,1024},{36,1024,256},
        {38,256,1024},{40,1024,1024},{42,1024,1024},{44,1024,1024},
        {46,2048,1024},{48,1024,1024},{50,1024,1024},{52,1024,512},
        {54,2048,3072} };

    char* wp = (char*)d_ws;
    auto alloc = [&](size_t bytes) -> char* {
        char* p = wp;
        wp += (bytes + 255) & ~(size_t)255;
        return p;
    };

    // 1) Convert + transpose all weights to bf16 [N,K] (~61 MB, L2-resident)
    bf16_t* Wt[NW];
    for (int i = 0; i < NW; ++i) {
        long elems = (long)ws[i].K * ws[i].N;
        Wt[i] = (bf16_t*)alloc(elems * sizeof(bf16_t));
        wcvt_kernel<<<dim3((unsigned)((elems + 255) / 256)), dim3(256), 0, stream>>>(
            (const float*)d_in[ws[i].idx], ws[i].K, ws[i].N, Wt[i]);
    }
    auto bptr = [&](int wi) { return (const float*)d_in[ws[wi].idx + 1]; };

    // scratch
    const int  CH = 4096;
    const long BT = (long)Bc * Tt;
    bf16_t* xcbf  = (bf16_t*)alloc((long)CH * Xd * sizeof(bf16_t));
    bf16_t* xt0   = (bf16_t*)alloc((long)CH * Hd * sizeof(bf16_t));
    bf16_t* xt1   = (bf16_t*)alloc((long)CH * Hd * sizeof(bf16_t));
    bf16_t* xfeat = (bf16_t*)alloc(BT * Hd * sizeof(bf16_t));   // 64 MB
    bf16_t* h     = (bf16_t*)alloc((long)Bc * Hd * sizeof(bf16_t));
    bf16_t* pf    = (bf16_t*)alloc((long)Bc * Hd * sizeof(bf16_t));
    bf16_t* ef    = (bf16_t*)alloc((long)Bc * Hd * sizeof(bf16_t));
    bf16_t* zf    = (bf16_t*)alloc((long)Bc * Hd * sizeof(bf16_t));
    bf16_t* t1    = (bf16_t*)alloc((long)Bc * Hd * sizeof(bf16_t));
    bf16_t* t2    = (bf16_t*)alloc((long)Bc * Hd * sizeof(bf16_t));
    bf16_t* t1b   = (bf16_t*)alloc((long)Bc * Hd * sizeof(bf16_t));
    bf16_t* t2b   = (bf16_t*)alloc((long)Bc * Hd * sizeof(bf16_t));
    bf16_t* zsb   = (bf16_t*)alloc((long)Bc * Zd * sizeof(bf16_t));
    float*  gi    = (float*)alloc((long)Bc * 3 * Hd * sizeof(float));

    const bf16_t* NB = nullptr;
    const int KBIG = 1 << 30;

    // single-problem launch (optionally with concat A)
    auto gemm1 = [&](const bf16_t* A0, long lda0, const bf16_t* A1, long lda1, int K0,
                     int wi, float* Cf, long ldcf, bf16_t* Cb, long ldcb,
                     int M, int act) {
        dim3 grid((unsigned)(ws[wi].N / 64), (unsigned)(M / 128), 1);
        gemm_bf16_wmma<<<grid, dim3(256), 0, stream>>>(
            A0, A0, lda0, A1, lda1, K0, Wt[wi], Wt[wi], ws[wi].K,
            bptr(wi), bptr(wi), Cf, Cf, ldcf, Cb, Cb, ldcb, act, act);
    };
    // dual-problem launch (mu/sigma pair): same shape, independent A/W/C/act
    auto gemm2 = [&](const bf16_t* Aa, const bf16_t* Ab, long lda0,
                     int wa, int wb,
                     float* CfA, float* CfB, long ldcf,
                     bf16_t* CbA, bf16_t* CbB, long ldcb,
                     int M, int actA, int actB) {
        dim3 grid((unsigned)(ws[wa].N / 64), (unsigned)(M / 128), 2);
        gemm_bf16_wmma<<<grid, dim3(256), 0, stream>>>(
            Aa, Ab, lda0, NB, 0, KBIG, Wt[wa], Wt[wb], ws[wa].K,
            bptr(wa), bptr(wb), CfA, CfB, ldcf, CbA, CbB, ldcb, actA, actB);
    };

    // 2) h-independent x_feats MLP over all B*T rows (parallel, chunked)
    for (long c = 0; c < BT / CH; ++c) {
        f2bf_kernel<<<dim3((unsigned)((CH * (long)Xd) / 256)), dim3(256), 0, stream>>>(
            x + c * CH * Xd, xcbf, (long)CH * Xd);
        gemm1(xcbf, Xd, NB, 0, KBIG, W_XF0, nullptr, 0, xt0, Hd, CH, ACT_RELU);
        gemm1(xt0,  Hd, NB, 0, KBIG, W_XF1, nullptr, 0, xt1, Hd, CH, ACT_RELU);
        gemm1(xt1,  Hd, NB, 0, KBIG, W_XF2, nullptr, 0, xt0, Hd, CH, ACT_RELU);
        gemm1(xt0,  Hd, NB, 0, KBIG, W_XF3, nullptr, 0,
              xfeat + c * CH * Hd, Hd, CH, ACT_RELU);
    }

    // output sections: (p_mu, p_sigma, z_mu, z_sigma, dec), each [B,T,*]
    float* dout   = (float*)d_out;
    const long BTZ = (long)Bc * Tt * Zd;
    float* o_pmu  = dout;
    float* o_psig = dout + BTZ;
    float* o_zmu  = dout + 2 * BTZ;
    float* o_zsig = dout + 3 * BTZ;
    float* o_dec  = dout + 4 * BTZ;
    const long ldZ  = (long)Tt * Zd;
    const long ldX  = (long)Tt * Xd;
    const long ldxf = (long)Tt * Hd;

    zerobf_kernel<<<dim3((Bc * Hd) / 256), dim3(256), 0, stream>>>(h, (long)Bc * Hd);

    // 3) sequential scan over T (weights hot in L2)
    for (int t = 0; t < Tt; ++t) {
        const bf16_t* xf_t = xfeat + (long)t * Hd;

        // prior: pf then paired mu/sigma chains
        gemm1(h, Hd, NB, 0, KBIG, W_PF, nullptr, 0, pf, Hd, Bc, ACT_RELU);
        gemm2(pf, pf, Hd, W_PM0, W_PS0, nullptr, nullptr, 0, t1, t1b, Hd,
              Bc, ACT_RELU, ACT_RELU);
        gemm2(t1, t1b, Hd, W_PM1, W_PS1, nullptr, nullptr, 0, t2, t2b, Hd,
              Bc, ACT_RELU, ACT_RELU);
        gemm2(t2, t2b, Hd, W_PM2, W_PS2,
              o_pmu + (long)t * Zd, o_psig + (long)t * Zd, ldZ,
              nullptr, nullptr, 0, Bc, ACT_NONE, ACT_SP);

        // encoder: concat(x_feat, h) then paired mu/sigma chains
        gemm1(xf_t, ldxf, h, Hd, Hd, W_EF, nullptr, 0, ef, Hd, Bc, ACT_RELU);
        gemm2(ef, ef, Hd, W_EM0, W_ES0, nullptr, nullptr, 0, t1, t1b, Hd,
              Bc, ACT_RELU, ACT_RELU);
        gemm2(t1, t1b, Hd, W_EM1, W_ES1, nullptr, nullptr, 0, t2, t2b, Hd,
              Bc, ACT_RELU, ACT_RELU);
        gemm2(t2, t2b, Hd, W_EM2, W_ES2,
              o_zmu + (long)t * Zd, o_zsig + (long)t * Zd, ldZ,
              nullptr, nullptr, 0, Bc, ACT_NONE, ACT_SP);

        // reparameterization
        zsample_kernel<<<dim3((Bc * Zd) / 256), dim3(256), 0, stream>>>(
            o_zmu + (long)t * Zd, o_zsig + (long)t * Zd, ldZ,
            eps + (long)t * Zd, ldZ, zsb, Bc * Zd);

        // z_feats
        gemm1(zsb, Zd, NB, 0, KBIG, W_ZF0, nullptr, 0, t1, Hd, Bc, ACT_RELU);
        gemm1(t1,  Hd, NB, 0, KBIG, W_ZF1, nullptr, 0, t2, Hd, Bc, ACT_RELU);
        gemm1(t2,  Hd, NB, 0, KBIG, W_ZF2, nullptr, 0, t1, Hd, Bc, ACT_RELU);
        gemm1(t1,  Hd, NB, 0, KBIG, W_ZF3, nullptr, 0, zf, Hd, Bc, ACT_RELU);

        // decoder: concat(z_feat, h)
        gemm1(zf, Hd, h, Hd, Hd, W_D0, nullptr, 0, t1, Hd, Bc, ACT_RELU);
        gemm1(t1, Hd, NB, 0, KBIG, W_D1, nullptr, 0, t2, Hd, Bc, ACT_RELU);
        gemm1(t2, Hd, NB, 0, KBIG, W_D2, nullptr, 0, t1, Hd, Bc, ACT_RELU);
        gemm1(t1, Hd, NB, 0, KBIG, W_D3, o_dec + (long)t * Xd, ldX,
              nullptr, 0, Bc, ACT_NONE);

        // GRU (hx == 0): gi = concat(x_feat, z_feat) @ W_ih + b_ih, then gates
        gemm1(xf_t, ldxf, zf, Hd, Hd, W_IH, gi, 3 * Hd, nullptr, 0, Bc, ACT_NONE);
        gru_gate_kernel<<<dim3((Bc * Hd) / 256), dim3(256), 0, stream>>>(
            gi, b_hh, h, Bc * Hd);
    }
}
